// GCNLayer_35983236006066
// MI455X (gfx1250) — matrix operand order repaired and checked by reading, not compiled, not run on hardware
//
#include <hip/hip_runtime.h>
#include <hip/hip_fp16.h>

typedef __attribute__((ext_vector_type(16))) _Float16 v16h;
typedef __attribute__((ext_vector_type(8)))  float    v8f;

#define DIN  128
#define DOUT 64
#define SLOPE 0.01f

// ---------------------------------------------------------------------------
// K0: fold BatchNorm into the weight matrix AND pre-swizzle into the
// wave32 B-fragment order this code uses for v_wmma_f32_16x16x32_f16 (contiguous k):
//   chunk = (kstep*4 + ntile), lane = (K-hi-group)*16 + (col%16)
//   W2sw[((chunk*32 + lane)*16) + h] = scale_k * W[k,j]  where
//   k = kstep*32 + (lane>>4)*16 + h,  j = ntile*16 + (lane&15)
// Each lane's 32-byte fragment is then contiguous -> ds_load_b128 pairs.
// b2[j] = b[j] + sum_k (beta_k - mean_k*scale_k) * W[k,j]
// ---------------------------------------------------------------------------
__global__ void fold_kernel(const float* __restrict__ gamma, const float* __restrict__ beta,
                            const float* __restrict__ rmean, const float* __restrict__ rvar,
                            const float* __restrict__ W, const float* __restrict__ b,
                            _Float16* __restrict__ W2sw, float* __restrict__ b2) {
    int j  = threadIdx.x;            // 0..63 (output column)
    int nt = j >> 4;                 // n-tile
    int cl = j & 15;                 // column within n-tile
    float acc = b[j];
    for (int k = 0; k < DIN; ++k) {
        float s = gamma[k] * rsqrtf(rvar[k] + 1e-5f);
        float w = W[k * DOUT + j];
        int kki  = k >> 5;           // k-step (0..3)
        int krem = k & 31;
        int hig  = krem >> 4;        // K hi-group (0/1)
        int h    = krem & 15;        // half index within fragment
        int lane = hig * 16 + cl;
        W2sw[(((kki * 4 + nt) * 32 + lane) << 4) + h] = (_Float16)(s * w);
        acc += (beta[k] - rmean[k] * s) * w;
    }
    b2[j] = acc;
}

// ---------------------------------------------------------------------------
// K1: per-row reciprocal L2 norm. One wave32 per row, float4 loads.
// ---------------------------------------------------------------------------
__global__ __launch_bounds__(256) void rownorm_kernel(const float* __restrict__ H,
                                                      float* __restrict__ rnorm, int N) {
    int wave = threadIdx.x >> 5;
    int lane = threadIdx.x & 31;
    int row  = blockIdx.x * 8 + wave;
    if (row >= N) return;
    const float4* h4 = (const float4*)(H + (size_t)row * DIN);
    float4 v = h4[lane];                       // 32 lanes * 4 = 128 elems
    float s = v.x * v.x + v.y * v.y + v.z * v.z + v.w * v.w;
    #pragma unroll
    for (int o = 16; o > 0; o >>= 1) s += __shfl_xor(s, o, 32);
    if (lane == 0) rnorm[row] = 1.0f / fmaxf(sqrtf(s), 1e-12f);
}

// ---------------------------------------------------------------------------
// K2: X = rnorm .* (H @ W2) + b2 via v_wmma_f32_16x16x32_f16.
// One wave per 16-row tile; K=128 -> 4 k-steps; N=64 -> 4 n-tiles.
// W2 staged in LDS once per block (already in B-fragment order -> v16h loads).
// Per k-step: batch all 4 B fragments (8x ds_load_b128, one wait), then
// 4 back-to-back WMMAs into distinct accumulators (no D->A/B hazard).
// ---------------------------------------------------------------------------
__global__ __launch_bounds__(256) void gemm_wmma_kernel(const float* __restrict__ H,
                                                        const _Float16* __restrict__ W2sw,
                                                        const float* __restrict__ b2,
                                                        const float* __restrict__ rnorm,
                                                        float* __restrict__ X,
                                                        int n_tiles, int N) {
    __shared__ __align__(32) _Float16 sW[DIN * DOUT];   // 16384 B, B-fragment order
    __shared__ float sb[DOUT];

    int tid = threadIdx.x;
    // stage: contiguous 16 KB copy as uint4 (1024 x 16 B, 256 threads x 4)
    {
        const uint4* src = (const uint4*)W2sw;
        uint4* dst = (uint4*)sW;
        #pragma unroll
        for (int i = tid; i < (DIN * DOUT * 2 / 16); i += 256) dst[i] = src[i];
        if (tid < DOUT) sb[tid] = b2[tid];
    }
    __syncthreads();

    int wave = tid >> 5;
    int lane = tid & 31;
    int tile = blockIdx.x * 8 + wave;
    if (tile >= n_tiles) return;     // wave-uniform: EXEC stays all-1s for WMMA

    int row0 = tile * 16;
    int m    = lane & 15;            // A: row within tile / B,D: column within n-tile
    int hi   = lane >> 4;            // lane group
    int kbA  = hi * 16;              // A-fragment K base (contiguous: k = 16*hi + s, matches B)

    int arow = row0 + m;
    if (arow >= N) arow = N - 1;     // safety clamp (N is a multiple of 16 here)
    const float* hrow = H + (size_t)arow * DIN;

    const v16h* sWv = (const v16h*)sW;

    v8f acc0 = {}, acc1 = {}, acc2 = {}, acc3 = {};

    #pragma unroll
    for (int kki = 0; kki < 4; ++kki) {
        int kk = kki * 32;
        // ---- A fragment: two contiguous 8-float runs -> 4x float4 loads ----
        const float4* pA0 = (const float4*)(hrow + kk + kbA);
        const float4* pA1 = (const float4*)(hrow + kk + kbA + 8);
        float4 p0 = pA0[0], p1 = pA0[1];
        float4 p2 = pA1[0], p3 = pA1[1];
        v16h a;
        a[0]  = (_Float16)p0.x;  a[1]  = (_Float16)p0.y;
        a[2]  = (_Float16)p0.z;  a[3]  = (_Float16)p0.w;
        a[4]  = (_Float16)p1.x;  a[5]  = (_Float16)p1.y;
        a[6]  = (_Float16)p1.z;  a[7]  = (_Float16)p1.w;
        a[8]  = (_Float16)p2.x;  a[9]  = (_Float16)p2.y;
        a[10] = (_Float16)p2.z;  a[11] = (_Float16)p2.w;
        a[12] = (_Float16)p3.x;  a[13] = (_Float16)p3.y;
        a[14] = (_Float16)p3.z;  a[15] = (_Float16)p3.w;

        // ---- batch all 4 B fragments, then 4 back-to-back WMMAs ----
        v16h bf0 = sWv[(kki * 4 + 0) * 32 + lane];
        v16h bf1 = sWv[(kki * 4 + 1) * 32 + lane];
        v16h bf2 = sWv[(kki * 4 + 2) * 32 + lane];
        v16h bf3 = sWv[(kki * 4 + 3) * 32 + lane];

        acc0 = __builtin_amdgcn_wmma_f32_16x16x32_f16(false, a, false, bf0,
                                                      (short)0, acc0, false, false);
        acc1 = __builtin_amdgcn_wmma_f32_16x16x32_f16(false, a, false, bf1,
                                                      (short)0, acc1, false, false);
        acc2 = __builtin_amdgcn_wmma_f32_16x16x32_f16(false, a, false, bf2,
                                                      (short)0, acc2, false, false);
        acc3 = __builtin_amdgcn_wmma_f32_16x16x32_f16(false, a, false, bf3,
                                                      (short)0, acc3, false, false);
    }

    // ---- store: D layout lane L, vgpr j -> row j + hi*8, col (L&15) ----
    int mb = hi * 8;
    #pragma unroll
    for (int nt = 0; nt < 4; ++nt) {
        int col = nt * 16 + m;
        float bias = sb[col];
        v8f accv = (nt == 0) ? acc0 : (nt == 1) ? acc1 : (nt == 2) ? acc2 : acc3;
        #pragma unroll
        for (int j = 0; j < 8; ++j) {
            int r = row0 + mb + j;
            if (r < N) X[(size_t)r * DOUT + col] = rnorm[r] * accv[j] + bias;
        }
    }
}

// ---------------------------------------------------------------------------
// K3: zero the accumulator (d_out is poisoned by the harness).
// ---------------------------------------------------------------------------
__global__ void zero_kernel(float* __restrict__ out, int n) {
    int i = blockIdx.x * 256 + threadIdx.x;
    if (i < n) out[i] = 0.0f;
}

// ---------------------------------------------------------------------------
// K4: edge-parallel SpMM, 64 threads per edge, f32 atomics into L2-resident out.
// ---------------------------------------------------------------------------
__global__ __launch_bounds__(256) void spmm_kernel(const int* __restrict__ rows,
                                                   const int* __restrict__ cols,
                                                   const float* __restrict__ vals,
                                                   const float* __restrict__ X,
                                                   float* __restrict__ out, int E) {
    int e = blockIdx.x * 4 + (threadIdx.x >> 6);
    if (e >= E) return;
    int j = threadIdx.x & 63;
    int r = rows[e];
    int c = cols[e];
    float v = vals[e];
    float x = X[(size_t)c * DOUT + j];
    unsafeAtomicAdd(&out[(size_t)r * DOUT + j], v * x);
}

// ---------------------------------------------------------------------------
// K5: LeakyReLU in place.
// ---------------------------------------------------------------------------
__global__ void leaky_kernel(float* __restrict__ out, int n) {
    int i = blockIdx.x * 256 + threadIdx.x;
    if (i < n) {
        float v = out[i];
        out[i] = (v >= 0.0f) ? v : SLOPE * v;
    }
}

// ---------------------------------------------------------------------------
extern "C" void kernel_launch(void* const* d_in, const int* in_sizes, int n_in,
                              void* d_out, int out_size, void* d_ws, size_t ws_size,
                              hipStream_t stream) {
    (void)n_in; (void)out_size; (void)ws_size;
    const float* H     = (const float*)d_in[0];
    const int*   rows  = (const int*)  d_in[1];
    const int*   cols  = (const int*)  d_in[2];
    const float* vals  = (const float*)d_in[3];
    const float* gamma = (const float*)d_in[4];
    const float* beta  = (const float*)d_in[5];
    const float* rmean = (const float*)d_in[6];
    const float* rvar  = (const float*)d_in[7];
    const float* W     = (const float*)d_in[8];
    const float* b     = (const float*)d_in[9];
    float* out = (float*)d_out;

    int N = in_sizes[0] / DIN;
    int E = in_sizes[1];

    // workspace layout
    char* ws = (char*)d_ws;
    _Float16* W2sw  = (_Float16*)ws;                       // 16384 B (swizzled)
    float*    b2    = (float*)(ws + 16384);                // 256 B
    float*    rnorm = (float*)(ws + 16640);                // N*4 B
    size_t rn_bytes = (((size_t)N * 4) + 255) & ~(size_t)255;
    float*    X     = (float*)(ws + 16640 + rn_bytes);     // N*64*4 B

    fold_kernel<<<1, DOUT, 0, stream>>>(gamma, beta, rmean, rvar, W, b, W2sw, b2);

    rownorm_kernel<<<(N + 7) / 8, 256, 0, stream>>>(H, rnorm, N);

    int n_tiles = (N + 15) / 16;
    gemm_wmma_kernel<<<(n_tiles + 7) / 8, 256, 0, stream>>>(H, W2sw, b2, rnorm, X, n_tiles, N);

    int total = N * DOUT;
    zero_kernel<<<(total + 255) / 256, 256, 0, stream>>>(out, total);

    spmm_kernel<<<(E + 3) / 4, 256, 0, stream>>>(rows, cols, vals, X, out, E);

    leaky_kernel<<<(total + 255) / 256, 256, 0, stream>>>(out, total);
}
